// W8A16LinearStatic_21251498180870
// MI455X (gfx1250) — matrix operand order repaired and checked by reading, not compiled, not run on hardware
//
#include <hip/hip_runtime.h>
#include <stdint.h>

// Problem constants (reference: M,K,N = 512, 4096, 11008)
#define GM 512
#define GK 4096
#define GN 11008

#define BM 128          // output tile rows per workgroup
#define BN 128          // output tile cols per workgroup
#define BK 64           // K-slice staged in LDS per stage
#define NSTAGES (GK / BK)
#define LDS_STRIDE 72   // halves per row: 64 + 8 pad -> 144B, 16B aligned, conflict-free

typedef __attribute__((ext_vector_type(16))) _Float16 v16h;
typedef __attribute__((ext_vector_type(2)))  _Float16 v2h;
typedef __attribute__((ext_vector_type(2)))  __fp16   v2fp16;  // cvt_pkrtz result type
typedef __attribute__((ext_vector_type(8)))  float    v8f;

struct FragU { uint32_t u[8]; };   // 32 bytes == 16 f16

// f32 x2 -> packed f16 pair (a in low half, b in high half), single VALU op
__device__ __forceinline__ uint32_t pack2_f16(float a, float b) {
#if defined(__has_builtin) && __has_builtin(__builtin_amdgcn_cvt_pkrtz)
    v2fp16 r = __builtin_amdgcn_cvt_pkrtz(a, b);
    return __builtin_bit_cast(uint32_t, r);
#else
    v2h r;
    r.x = (_Float16)a;
    r.y = (_Float16)b;
    return __builtin_bit_cast(uint32_t, r);
#endif
}

// ---------------------------------------------------------------------------
// Pre-pass: rowsum_x[m] = sum_k x[m,k]  (for the per-channel offset term)
// ---------------------------------------------------------------------------
__global__ void rowsum_kernel(const float* __restrict__ x, float* __restrict__ rs) {
    const int row = blockIdx.x;
    const int t   = threadIdx.x;
    float s = 0.f;
    const float* xr = x + (size_t)row * GK;
    #pragma unroll
    for (int i = 0; i < GK / 256; ++i) s += xr[t + 256 * i];
    __shared__ float red[256];
    red[t] = s;
    __syncthreads();
    for (int off = 128; off > 0; off >>= 1) {
        if (t < off) red[t] += red[t + off];
        __syncthreads();
    }
    if (t == 0) rs[row] = red[0];
}

// ---------------------------------------------------------------------------
// Main GEMM: y[m,n] = scale[n] * (sum_k xh[m,k]*q[n,k] + off[n]*rowsum[m]) + bias[n]
// f16 WMMA 16x16x32, f32 accumulate. int8 weights are exact in f16.
// 256 threads = 8 waves (2x4 wave grid). Ping-pong LDS, one barrier per stage.
// ---------------------------------------------------------------------------
__global__ __launch_bounds__(256)
void w8a16_gemm_kernel(const float* __restrict__ x,
                       const int*   __restrict__ wq,
                       const float* __restrict__ scale,
                       const float* __restrict__ offneg,
                       const float* __restrict__ bias,
                       const float* __restrict__ rowsum,
                       float*       __restrict__ y) {
    __shared__ __align__(16) unsigned short sA[2][BM * LDS_STRIDE]; // x tiles, f16
    __shared__ __align__(16) unsigned short sB[2][BN * LDS_STRIDE]; // w tiles, f16

    const int tid  = threadIdx.x;
    const int lane = tid & 31;           // wave32
    const int wave = tid >> 5;           // 0..7
    const int wm   = wave >> 2;          // 0..1  -> 64 rows each
    const int wn   = wave & 3;           // 0..3  -> 32 cols each
    const int hi   = lane >> 4;          // half-wave select
    const int lr   = lane & 15;

    const int m0 = blockIdx.y * BM;
    const int n0 = blockIdx.x * BN;

    const float4* __restrict__ xv = (const float4*)x;
    const int4*   __restrict__ wv = (const int4*)wq;

    v8f acc[4][2] = {};                  // 4 m-subtiles x 2 n-subtiles of 16x16

    // Per-thread staging: 2048 float4-chunks per tile, 8 per thread.
    // 32-bit element offsets (float4 units) -> SGPR base + VGPR offset addressing.
    int srow[8], scol[8];
    uint32_t xo[8], wo[8];
    #pragma unroll
    for (int i = 0; i < 8; ++i) {
        const int c = tid + 256 * i;
        srow[i] = c >> 4;                // 0..127
        scol[i] = c & 15;                // 0..15 (float4 chunks)
        xo[i] = (uint32_t)(m0 + srow[i]) * (GK / 4) + scol[i];
        wo[i] = (uint32_t)(n0 + srow[i]) * (GK / 4) + scol[i];
    }

    // ---- prologue: fetch stage 0, publish into buffer 0 ----
    float4 ax[8];
    int4   bw[8];
    #pragma unroll
    for (int i = 0; i < 8; ++i) {
        ax[i] = xv[xo[i]];
        bw[i] = wv[wo[i]];
        xo[i] += BK / 4;
        wo[i] += BK / 4;
    }
    #pragma unroll
    for (int i = 0; i < 8; ++i) {
        uint2 pa;
        pa.x = pack2_f16(ax[i].x, ax[i].y);
        pa.y = pack2_f16(ax[i].z, ax[i].w);
        *(uint2*)&sA[0][srow[i] * LDS_STRIDE + scol[i] * 4] = pa;
        uint2 pb;
        pb.x = pack2_f16((float)bw[i].x, (float)bw[i].y);
        pb.y = pack2_f16((float)bw[i].z, (float)bw[i].w);
        *(uint2*)&sB[0][srow[i] * LDS_STRIDE + scol[i] * 4] = pb;
    }

    for (int kt = 0; kt < NSTAGES; ++kt) {
        const int cur = kt & 1;
        const int nxt = cur ^ 1;

        // ---- issue global loads for next stage (latency hidden behind WMMA) ----
        if (kt + 1 < NSTAGES) {
            #pragma unroll
            for (int i = 0; i < 8; ++i) {
                ax[i] = xv[xo[i]];
                bw[i] = wv[wo[i]];
                xo[i] += BK / 4;
                wo[i] += BK / 4;
            }
            // CDNA5 prefetch of stage kt+2 (global_prefetch_b8)
            __builtin_prefetch(&wv[wo[0]], 0, 1);
            __builtin_prefetch(&xv[xo[0]], 0, 1);
        }

        __syncthreads();  // stage kt published; stage kt-1 reads of buffer `nxt` done

        // ---- publish next stage into the other buffer (overlaps compute) ----
        if (kt + 1 < NSTAGES) {
            #pragma unroll
            for (int i = 0; i < 8; ++i) {
                uint2 pa;
                pa.x = pack2_f16(ax[i].x, ax[i].y);
                pa.y = pack2_f16(ax[i].z, ax[i].w);
                *(uint2*)&sA[nxt][srow[i] * LDS_STRIDE + scol[i] * 4] = pa;
                uint2 pb;
                pb.x = pack2_f16((float)bw[i].x, (float)bw[i].y);
                pb.y = pack2_f16((float)bw[i].z, (float)bw[i].w);
                *(uint2*)&sB[nxt][srow[i] * LDS_STRIDE + scol[i] * 4] = pb;
            }
        }

        // ---- compute stage kt: 2 k-steps of 32 ----
        const unsigned short* __restrict__ A = sA[cur];
        const unsigned short* __restrict__ B = sB[cur];
        #pragma unroll
        for (int kk = 0; kk < 2; ++kk) {
            const int kb = kk * 32;

            // A fragments: 16-bit A 16x32 layout.
            // lane<16: K 0..7 in V0..3, K 16..23 in V4..7; lane>=16: +8.
            FragU af[4];
            #pragma unroll
            for (int sm = 0; sm < 4; ++sm) {
                const int row = wm * 64 + sm * 16 + lr;
                const uint4 q0 = *(const uint4*)&A[row * LDS_STRIDE + kb + hi * 8];
                const uint4 q1 = *(const uint4*)&A[row * LDS_STRIDE + kb + 16 + hi * 8];
                af[sm].u[0] = q0.x; af[sm].u[1] = q0.y; af[sm].u[2] = q0.z; af[sm].u[3] = q0.w;
                af[sm].u[4] = q1.x; af[sm].u[5] = q1.y; af[sm].u[6] = q1.z; af[sm].u[7] = q1.w;
            }

            // B fragments: column-per-lane; lane<16: K 0..7 in V0..3, K 16..23 in V4..7; lane>=16: +8.
            FragU bfg[2];
            #pragma unroll
            for (int sn = 0; sn < 2; ++sn) {
                const int col = wn * 32 + sn * 16 + lr;
                const uint4 q0 = *(const uint4*)&B[col * LDS_STRIDE + kb + hi * 8];
                const uint4 q1 = *(const uint4*)&B[col * LDS_STRIDE + kb + 16 + hi * 8];
                bfg[sn].u[0] = q0.x; bfg[sn].u[1] = q0.y; bfg[sn].u[2] = q0.z; bfg[sn].u[3] = q0.w;
                bfg[sn].u[4] = q1.x; bfg[sn].u[5] = q1.y; bfg[sn].u[6] = q1.z; bfg[sn].u[7] = q1.w;
            }

            #pragma unroll
            for (int sm = 0; sm < 4; ++sm) {
                #pragma unroll
                for (int sn = 0; sn < 2; ++sn) {
                    acc[sm][sn] = __builtin_amdgcn_wmma_f32_16x16x32_f16(
                        /*neg_a=*/false, __builtin_bit_cast(v16h, af[sm]),
                        /*neg_b=*/false, __builtin_bit_cast(v16h, bfg[sn]),
                        /*c_mod=*/(short)0, acc[sm][sn],
                        /*reuse_a=*/false, /*reuse_b=*/false);
                }
            }
        }
    }

    // ---- epilogue: y = scale*(acc + off*rowsum) + bias ----
    // C/D layout: VGPR r, lanes 0-15 -> M = base + r, lanes 16-31 -> M = base + 8 + r.
    #pragma unroll
    for (int sn = 0; sn < 2; ++sn) {
        const int n = n0 + wn * 32 + sn * 16 + lr;
        const float s = scale[n];
        const float o = offneg[n];
        const float b = bias[n];
        #pragma unroll
        for (int sm = 0; sm < 4; ++sm) {
            const int mbase = m0 + wm * 64 + sm * 16 + hi * 8;
            #pragma unroll
            for (int r = 0; r < 8; ++r) {
                const int m = mbase + r;
                const float v = s * (acc[sm][sn][r] + o * rowsum[m]) + b;
                y[(size_t)m * GN + n] = v;
            }
        }
    }
}

extern "C" void kernel_launch(void* const* d_in, const int* in_sizes, int n_in,
                              void* d_out, int out_size, void* d_ws, size_t ws_size,
                              hipStream_t stream) {
    const float* x      = (const float*)d_in[0];
    const int*   wq     = (const int*)d_in[1];
    const float* scale  = (const float*)d_in[2];
    const float* offneg = (const float*)d_in[3];
    const float* bias   = (const float*)d_in[4];
    float*       y      = (float*)d_out;
    float*       rowsum = (float*)d_ws;   // 512 floats

    rowsum_kernel<<<GM, 256, 0, stream>>>(x, rowsum);

    dim3 grid(GN / BN, GM / BM);          // 86 x 4
    w8a16_gemm_kernel<<<grid, 256, 0, stream>>>(x, wq, scale, offneg, bias, rowsum, y);
}
